// Fine_Grain_Layer_34093450395752
// MI455X (gfx1250) — compile-verified, matrix operand order checked by reading, not run on hardware
//
#include <hip/hip_runtime.h>
#include <hip/hip_bf16.h>

typedef __attribute__((ext_vector_type(16))) _Float16 v16h;
typedef __attribute__((ext_vector_type(4)))  _Float16 v4h;
typedef __attribute__((ext_vector_type(8)))  float    v8f;
typedef __attribute__((ext_vector_type(4)))  float    f4;

#define NN 8192
#define EE 262144
#define DD 128

// D = A(16x32 f16) * B(32x16 f16) + C(16x16 f32)
__device__ __forceinline__ v8f wmma16(v16h a, v16h b, v8f c) {
  return __builtin_amdgcn_wmma_f32_16x16x32_f16(false, a, false, b, (short)0, c, false, false);
}

// Loads an A- or B-fragment for v_wmma_f32_16x16x32_f16 from a row-major f16
// buffer (row = M for A / N-col for transposed-B weights), row stride `ld`.
// ISA layout: lane L (0..15) -> row L, k in {8h..8h+7} U {16+8h..16+8h+7},
// h = L>>4.  Both k-chunks are contiguous -> two 16-byte loads.
__device__ __forceinline__ v16h load_frag(const _Float16* base, int ld, int k0, int lane) {
  const _Float16* p = base + (size_t)(lane & 15) * (size_t)ld + k0 + 8 * (lane >> 4);
  v16h out;
  uint4* o = reinterpret_cast<uint4*>(&out);
  o[0] = *reinterpret_cast<const uint4*>(p);
  o[1] = *reinterpret_cast<const uint4*>(p + 16);
  return out;
}

__device__ __forceinline__ void store4h(_Float16* dst, f4 v) {
  *reinterpret_cast<v4h*>(dst) = (v4h){(_Float16)v.x, (_Float16)v.y, (_Float16)v.z, (_Float16)v.w};
}

__device__ __forceinline__ float lrelu(float x) { return x > 0.f ? x : 0.01f * x; }

__device__ __forceinline__ float rmax16(float v) {
  v = fmaxf(v, __shfl_xor(v, 1, 32));
  v = fmaxf(v, __shfl_xor(v, 2, 32));
  v = fmaxf(v, __shfl_xor(v, 4, 32));
  v = fmaxf(v, __shfl_xor(v, 8, 32));
  return v;
}
__device__ __forceinline__ float rsum16(float v) {
  v += __shfl_xor(v, 1, 32);
  v += __shfl_xor(v, 2, 32);
  v += __shfl_xor(v, 4, 32);
  v += __shfl_xor(v, 8, 32);
  return v;
}

// ---------------- utility kernels ----------------
__global__ void zero_kernel(float* __restrict__ p, int n) {
  int i = blockIdx.x * 256 + threadIdx.x;
  if (i < n) p[i] = 0.f;
}

// dst[n*Kpad + k] = f16(src[k*N + n]); zero-pad K..Kpad (transposed weights)
__global__ void convw_kernel(const float* __restrict__ src, _Float16* __restrict__ dst,
                             int K, int N, int Kpad) {
  int i = blockIdx.x * 256 + threadIdx.x;
  if (i >= N * Kpad) return;
  int n = i / Kpad, k = i - n * Kpad;
  dst[i] = (k < K) ? (_Float16)src[(size_t)k * N + n] : (_Float16)0.f;
}

__global__ void coords_kernel(const float* __restrict__ orig, const float* __restrict__ coords,
                              const float* __restrict__ xup, const float* __restrict__ cnt,
                              float* __restrict__ out) {
  int i = blockIdx.x * 256 + threadIdx.x;
  if (i >= NN * 3) return;
  int n = i / 3;
  float cdenom = fmaxf(cnt[n], 1.f);
  out[i] = 0.25f * orig[i] + 0.75f * coords[i] + xup[i] / cdenom;
}

// ---------------- QKV projection: out = act(h @ W) as f16 ----------------
__global__ __launch_bounds__(256) void lin_kernel(
    const float* __restrict__ h, const _Float16* __restrict__ WT,
    _Float16* __restrict__ out, int act, int trans)
{
  __shared__ __align__(16) _Float16 ht[64][128];
  const int tid = threadIdx.x, lane = tid & 31, wave = tid >> 5;
  const int r0 = blockIdx.x * 64;
  for (int idx = tid; idx < 64 * 32; idx += 256) {
    const int r = idx >> 5, j4 = (idx & 31) * 4;
    f4 v = *reinterpret_cast<const f4*>(&h[(size_t)(r0 + r) * 128 + j4]);
    store4h(&ht[r][j4], v);
  }
  __syncthreads();
  const int rt = wave & 3, ct = wave >> 2, col0 = ct * 64;
  const int c = lane & 15, hh = lane >> 4;
  v8f acc[4];
  for (int t = 0; t < 4; ++t) for (int i = 0; i < 8; ++i) acc[t][i] = 0.f;
  for (int kk = 0; kk < 128; kk += 32) {
    v16h a = load_frag(&ht[rt * 16][0], 128, kk, lane);
    for (int t = 0; t < 4; ++t)
      acc[t] = wmma16(a, load_frag(WT + (size_t)(col0 + 16 * t) * 128, 128, kk, lane), acc[t]);
  }
  for (int t = 0; t < 4; ++t) {
    const int col = col0 + 16 * t + c;
    for (int i = 0; i < 8; ++i) {
      const int row = r0 + rt * 16 + i + 8 * hh;
      float v = acc[t][i];
      if (act) v = lrelu(v);
      if (trans) out[(size_t)col * NN + row] = (_Float16)v;   // V stored transposed
      else       out[(size_t)row * 128 + col] = (_Float16)v;
    }
  }
}

// ---------------- fused edge branch ----------------
__global__ __launch_bounds__(256) void edge_kernel(
    const float* __restrict__ coords, const float* __restrict__ h,
    const float* __restrict__ efeat, const int* __restrict__ src,
    const int* __restrict__ dst,
    const _Float16* __restrict__ W1T, const float* __restrict__ b1,
    const _Float16* __restrict__ W2T, const float* __restrict__ b2,
    const _Float16* __restrict__ CW1T, const float* __restrict__ cb1,
    const float* __restrict__ cw2, const float* __restrict__ cb2,
    float* __restrict__ aggr_sum, float* __restrict__ xup_sum,
    float* __restrict__ cnt)
{
  __shared__ __align__(16) _Float16 zt[64][352];  // z rows, padded 335->352
  __shared__ __align__(16) _Float16 mt[64][128];  // layer-1 act, then msg2
  __shared__ float xrel[64][3];
  __shared__ float coefsum[64];
  __shared__ int   edst[64];

  const int tid = threadIdx.x, lane = tid & 31, wave = tid >> 5;
  const int e0 = blockIdx.x * 64;

  { // phase 1: gather + rbf -> z tile (f16).  h gathers hit L2 (h is 4MB,
    // L2-resident); efeat is a 64MB one-shot stream -> non-temporal.
    const int el = tid >> 2, sub = tid & 3;
    const int e = e0 + el;
    const int s = src[e], d = dst[e];
    for (int jj = 0; jj < 8; ++jj) {
      const int j = sub * 32 + jj * 4;
      f4 vs = *reinterpret_cast<const f4*>(&h[(size_t)s * DD + j]);
      f4 vd = *reinterpret_cast<const f4*>(&h[(size_t)d * DD + j]);
      store4h(&zt[el][j], vs);
      store4h(&zt[el][DD + j], vd);
    }
    for (int jj = 0; jj < 4; ++jj) {
      const int j = sub * 16 + jj * 4;
      f4 ve = __builtin_nontemporal_load(reinterpret_cast<const f4*>(&efeat[(size_t)e * 64 + j]));
      store4h(&zt[el][256 + j], ve);
    }
    if (sub == 0) {
      float x0 = coords[s * 3 + 0] - coords[d * 3 + 0];
      float x1 = coords[s * 3 + 1] - coords[d * 3 + 1];
      float x2 = coords[s * 3 + 2] - coords[d * 3 + 2];
      xrel[el][0] = x0; xrel[el][1] = x1; xrel[el][2] = x2;
      float d2 = x0 * x0 + x1 * x1 + x2 * x2;
      float sig = 1.0f;
      for (int k = 0; k < 15; ++k) { zt[el][320 + k] = (_Float16)__expf(-d2 / sig); sig *= 1.5f; }
      for (int k = 335; k < 352; ++k) zt[el][k] = (_Float16)0.f;
      edst[el] = d;
      coefsum[el] = 0.f;
      atomicAdd(&cnt[d], 1.0f);
    }
  }
  __syncthreads();

  const int rt = wave & 3, ct = wave >> 2, col0 = ct * 64;
  const int c = lane & 15, hh = lane >> 4;
  v8f acc[4];

  // GEMM1: [64x352] @ [352x128]
  for (int t = 0; t < 4; ++t) for (int i = 0; i < 8; ++i) acc[t][i] = 0.f;
  for (int kk = 0; kk < 352; kk += 32) {
    v16h a = load_frag(&zt[rt * 16][0], 352, kk, lane);
    for (int t = 0; t < 4; ++t)
      acc[t] = wmma16(a, load_frag(W1T + (size_t)(col0 + 16 * t) * 352, 352, kk, lane), acc[t]);
  }
  for (int t = 0; t < 4; ++t) {
    const int col = col0 + 16 * t + c;
    const float bias = b1[col];
    for (int i = 0; i < 8; ++i)
      mt[rt * 16 + i + 8 * hh][col] = (_Float16)lrelu(acc[t][i] + bias);
  }
  __syncthreads();

  // GEMM2: msg2 = act1 @ W2 + b2
  for (int t = 0; t < 4; ++t) for (int i = 0; i < 8; ++i) acc[t][i] = 0.f;
  for (int kk = 0; kk < 128; kk += 32) {
    v16h a = load_frag(&mt[rt * 16][0], 128, kk, lane);
    for (int t = 0; t < 4; ++t)
      acc[t] = wmma16(a, load_frag(W2T + (size_t)(col0 + 16 * t) * 128, 128, kk, lane), acc[t]);
  }
  float msg2[4][8];
  for (int t = 0; t < 4; ++t) {
    const float bias = b2[col0 + 16 * t + c];
    for (int i = 0; i < 8; ++i) msg2[t][i] = acc[t][i] + bias;
  }
  __syncthreads();  // everyone done reading layer-1 mt
  for (int t = 0; t < 4; ++t) {
    const int col = col0 + 16 * t + c;
    for (int i = 0; i < 8; ++i) {
      const int row = rt * 16 + i + 8 * hh;
      mt[row][col] = (_Float16)msg2[t][i];
      atomicAdd(&aggr_sum[(size_t)edst[row] * DD + col], msg2[t][i]);  // segment sum
    }
  }
  __syncthreads();

  // coef MLP: t = lrelu(msg2 @ cW1 + cb1); coef = t . cw2 + cb2
  for (int t = 0; t < 4; ++t) for (int i = 0; i < 8; ++i) acc[t][i] = 0.f;
  for (int kk = 0; kk < 128; kk += 32) {
    v16h a = load_frag(&mt[rt * 16][0], 128, kk, lane);
    for (int t = 0; t < 4; ++t)
      acc[t] = wmma16(a, load_frag(CW1T + (size_t)(col0 + 16 * t) * 128, 128, kk, lane), acc[t]);
  }
  for (int t = 0; t < 4; ++t) {
    const int col = col0 + 16 * t + c;
    const float bias = cb1[col], w2 = cw2[col];
    for (int i = 0; i < 8; ++i)
      atomicAdd(&coefsum[rt * 16 + i + 8 * hh], lrelu(acc[t][i] + bias) * w2);
  }
  __syncthreads();

  if (tid < 64) {
    const float coef = coefsum[tid] + cb2[0];
    const int d = edst[tid];
    atomicAdd(&xup_sum[d * 3 + 0], xrel[tid][0] * coef);
    atomicAdd(&xup_sum[d * 3 + 1], xrel[tid][1] * coef);
    atomicAdd(&xup_sum[d * 3 + 2], xrel[tid][2] * coef);
  }
}

// ---------------- flash cross-attention ----------------
// Each wave owns 16 Q rows end-to-end; P-tile bounce buffer is per-wave
// private LDS, so NO block barriers are needed inside the key loop (same-wave
// LDS ordering is enforced by DScnt waits).  mask is a 256MB one-shot stream:
// non-temporal loads keep K/V/weights resident in L2, and a prefetch two
// tiles ahead keeps the HBM stream in front of the WMMA pipeline.
__global__ __launch_bounds__(256) void attn_kernel(
    const _Float16* __restrict__ Qh, const _Float16* __restrict__ Kh,
    const _Float16* __restrict__ VT, const float* __restrict__ mask,
    float* __restrict__ crossA)
{
  __shared__ __align__(16) _Float16 pt[8][16][32];  // per-wave P tile
  const int tid = threadIdx.x, lane = tid & 31, wave = tid >> 5;
  const int qr0 = blockIdx.x * 128 + wave * 16;     // 16 Q rows per wave
  const int c = lane & 15, hh = lane >> 4;

  v16h qf[4];
  for (int k = 0; k < 4; ++k) qf[k] = load_frag(Qh + (size_t)qr0 * 128, 128, k * 32, lane);

  v8f o[8];
  for (int t = 0; t < 8; ++t) for (int i = 0; i < 8; ++i) o[t][i] = 0.f;
  float mrun[8], lrun[8];
  for (int i = 0; i < 8; ++i) { mrun[i] = -1e30f; lrun[i] = 0.f; }

  for (int jb = 0; jb < NN; jb += 32) {
    if (jb + 64 < NN)  // one 128B line per row, 2 tiles ahead
      __builtin_prefetch(&mask[(size_t)(qr0 + (lane & 15)) * NN + jb + 64], 0, 1);

    v8f s0, s1;
    for (int i = 0; i < 8; ++i) { s0[i] = 0.f; s1[i] = 0.f; }
    for (int k = 0; k < 4; ++k) {
      s0 = wmma16(qf[k], load_frag(Kh + (size_t)jb * 128, 128, k * 32, lane), s0);
      s1 = wmma16(qf[k], load_frag(Kh + (size_t)(jb + 16) * 128, 128, k * 32, lane), s1);
    }
    for (int i = 0; i < 8; ++i) {
      const size_t row = (size_t)(qr0 + i + 8 * hh);
      const float m0 = __builtin_nontemporal_load(&mask[row * NN + jb + c]);
      const float m1 = __builtin_nontemporal_load(&mask[row * NN + jb + 16 + c]);
      const float v0 = m0 * s0[i] - 1000.f * (1.f - m0);
      const float v1 = m1 * s1[i] - 1000.f * (1.f - m1);
      const float tmax = rmax16(fmaxf(v0, v1));
      const float mnew = fmaxf(mrun[i], tmax);
      const float scale = __expf(mrun[i] - mnew);
      const float p0 = __expf(v0 - mnew);
      const float p1 = __expf(v1 - mnew);
      lrun[i] = lrun[i] * scale + rsum16(p0 + p1);
      mrun[i] = mnew;
      for (int t = 0; t < 8; ++t) o[t][i] *= scale;
      pt[wave][i + 8 * hh][c]      = (_Float16)p0;
      pt[wave][i + 8 * hh][16 + c] = (_Float16)p1;
    }
    // same-wave LDS RAW: hardware DScnt ordering (compiler inserts the wait)
    v16h pf = load_frag(&pt[wave][0][0], 32, 0, lane);
    for (int t = 0; t < 8; ++t)
      o[t] = wmma16(pf, load_frag(VT + (size_t)(16 * t) * NN + jb, NN, 0, lane), o[t]);
  }
  for (int t = 0; t < 8; ++t)
    for (int i = 0; i < 8; ++i) {
      const size_t row = (size_t)(qr0 + i + 8 * hh);
      crossA[row * 128 + 16 * t + c] = o[t][i] / lrun[i];
    }
}

// ---------------- node MLP with skip ----------------
__global__ __launch_bounds__(256) void nodemlp_kernel(
    const float* __restrict__ h, const float* __restrict__ aggr_sum,
    const float* __restrict__ cnt, const float* __restrict__ cross, int use_cross,
    const float* __restrict__ orig_feat,
    const _Float16* __restrict__ W1T, const float* __restrict__ b1,
    const _Float16* __restrict__ W2T, const float* __restrict__ b2,
    float* __restrict__ out)
{
  __shared__ __align__(16) _Float16 it[32][512];
  __shared__ __align__(16) _Float16 mt2[32][128];
  const int tid = threadIdx.x, lane = tid & 31, wave = tid >> 5;
  const int n0 = blockIdx.x * 32;
  for (int idx = tid; idx < 32 * 128; idx += 256) {
    const int nl = idx >> 7;
    const int j4 = (idx & 127) * 4;
    const size_t n = (size_t)(n0 + nl);
    f4 v;
    if (j4 < 128) {
      v = *reinterpret_cast<const f4*>(&h[n * 128 + j4]);
    } else if (j4 < 256) {
      v = *reinterpret_cast<const f4*>(&aggr_sum[n * 128 + (j4 - 128)]);
      const float inv = 1.f / fmaxf(cnt[n], 1.f);
      v.x *= inv; v.y *= inv; v.z *= inv; v.w *= inv;
    } else if (j4 < 384) {
      if (use_cross) v = *reinterpret_cast<const f4*>(&cross[n * 128 + (j4 - 256)]);
      else { v.x = 0.f; v.y = 0.f; v.z = 0.f; v.w = 0.f; }
    } else {
      v = *reinterpret_cast<const f4*>(&orig_feat[n * 128 + (j4 - 384)]);
    }
    store4h(&it[nl][j4], v);
  }
  __syncthreads();
  const int rt = wave & 1, ct = wave >> 1;
  const int c = lane & 15, hh = lane >> 4;
  v8f acc[2];
  for (int t = 0; t < 2; ++t) for (int i = 0; i < 8; ++i) acc[t][i] = 0.f;
  for (int kk = 0; kk < 512; kk += 32) {
    v16h a = load_frag(&it[rt * 16][0], 512, kk, lane);
    for (int t = 0; t < 2; ++t)
      acc[t] = wmma16(a, load_frag(W1T + (size_t)(ct * 32 + 16 * t) * 512, 512, kk, lane), acc[t]);
  }
  for (int t = 0; t < 2; ++t) {
    const int col = ct * 32 + 16 * t + c;
    const float bias = b1[col];
    for (int i = 0; i < 8; ++i)
      mt2[rt * 16 + i + 8 * hh][col] = (_Float16)lrelu(acc[t][i] + bias);
  }
  __syncthreads();
  for (int t = 0; t < 2; ++t) for (int i = 0; i < 8; ++i) acc[t][i] = 0.f;
  for (int kk = 0; kk < 128; kk += 32) {
    v16h a = load_frag(&mt2[rt * 16][0], 128, kk, lane);
    for (int t = 0; t < 2; ++t)
      acc[t] = wmma16(a, load_frag(W2T + (size_t)(ct * 32 + 16 * t) * 128, 128, kk, lane), acc[t]);
  }
  for (int t = 0; t < 2; ++t) {
    const int col = ct * 32 + 16 * t + c;
    const float bias = b2[col];
    for (int i = 0; i < 8; ++i) {
      const size_t row = (size_t)(n0 + rt * 16 + i + 8 * hh);
      out[row * 128 + col] = 0.5f * (acc[t][i] + bias) + 0.5f * h[row * 128 + col];
    }
  }
}

// ---------------- host launch ----------------
extern "C" void kernel_launch(void* const* d_in, const int* in_sizes, int n_in,
                              void* d_out, int out_size, void* d_ws, size_t ws_size,
                              hipStream_t stream) {
  (void)in_sizes; (void)n_in; (void)out_size; (void)ws_size;
  const float* coords_A = (const float*)d_in[0];
  const float* h_A      = (const float*)d_in[1];
  const float* ofeat_A  = (const float*)d_in[2];
  const float* ocrd_A   = (const float*)d_in[3];
  const float* efeat_A  = (const float*)d_in[4];
  const float* coords_B = (const float*)d_in[5];
  const float* h_B      = (const float*)d_in[6];
  const float* ofeat_B  = (const float*)d_in[7];
  const float* ocrd_B   = (const float*)d_in[8];
  const float* efeat_B  = (const float*)d_in[9];
  const float* mask     = (const float*)d_in[10];
  const int* srcA = (const int*)d_in[11];
  const int* dstA = (const int*)d_in[12];
  const int* srcB = (const int*)d_in[13];
  const int* dstB = (const int*)d_in[14];
  const float* eA_W1 = (const float*)d_in[15]; const float* eA_b1 = (const float*)d_in[16];
  const float* eA_W2 = (const float*)d_in[17]; const float* eA_b2 = (const float*)d_in[18];
  const float* eB_W1 = (const float*)d_in[19]; const float* eB_b1 = (const float*)d_in[20];
  const float* eB_W2 = (const float*)d_in[21]; const float* eB_b2 = (const float*)d_in[22];
  const float* WQ = (const float*)d_in[23];
  const float* WK = (const float*)d_in[24];
  const float* WV = (const float*)d_in[25];
  const float* nA_W1 = (const float*)d_in[26]; const float* nA_b1 = (const float*)d_in[27];
  const float* nA_W2 = (const float*)d_in[28]; const float* nA_b2 = (const float*)d_in[29];
  const float* nB_W1 = (const float*)d_in[30]; const float* nB_b1 = (const float*)d_in[31];
  const float* nB_W2 = (const float*)d_in[32]; const float* nB_b2 = (const float*)d_in[33];
  const float* cA_W1 = (const float*)d_in[34]; const float* cA_b1 = (const float*)d_in[35];
  const float* cA_W2 = (const float*)d_in[36]; const float* cA_b2 = (const float*)d_in[37];
  const float* cB_W1 = (const float*)d_in[38]; const float* cB_b1 = (const float*)d_in[39];
  const float* cB_W2 = (const float*)d_in[40]; const float* cB_b2 = (const float*)d_in[41];

  char* ws = (char*)d_ws;
  size_t off = 0;
  auto alloc = [&](size_t bytes) -> void* {
    void* p = ws + off; off += (bytes + 255) & ~(size_t)255; return p;
  };
  _Float16* W1A_T  = (_Float16*)alloc(352 * 128 * 2);
  _Float16* W1B_T  = (_Float16*)alloc(352 * 128 * 2);
  _Float16* W2A_T  = (_Float16*)alloc(128 * 128 * 2);
  _Float16* W2B_T  = (_Float16*)alloc(128 * 128 * 2);
  _Float16* CW1A_T = (_Float16*)alloc(128 * 128 * 2);
  _Float16* CW1B_T = (_Float16*)alloc(128 * 128 * 2);
  _Float16* WQ_T   = (_Float16*)alloc(128 * 128 * 2);
  _Float16* WK_T   = (_Float16*)alloc(128 * 128 * 2);
  _Float16* WV_T   = (_Float16*)alloc(128 * 128 * 2);
  _Float16* NW1A_T = (_Float16*)alloc(512 * 128 * 2);
  _Float16* NW1B_T = (_Float16*)alloc(512 * 128 * 2);
  _Float16* NW2A_T = (_Float16*)alloc(128 * 128 * 2);
  _Float16* NW2B_T = (_Float16*)alloc(128 * 128 * 2);
  _Float16* Qh = (_Float16*)alloc((size_t)NN * 128 * 2);
  _Float16* Kh = (_Float16*)alloc((size_t)NN * 128 * 2);
  _Float16* VT = (_Float16*)alloc((size_t)128 * NN * 2);
  float* crossA = (float*)alloc((size_t)NN * 128 * 4);
  // contiguous accumulator region (zeroed each launch)
  float* aggrA = (float*)alloc((size_t)NN * 128 * 4);
  float* aggrB = (float*)alloc((size_t)NN * 128 * 4);
  float* xupA  = (float*)alloc((size_t)NN * 3 * 4);
  float* xupB  = (float*)alloc((size_t)NN * 3 * 4);
  float* cntA  = (float*)alloc((size_t)NN * 4);
  float* cntB  = (float*)alloc((size_t)NN * 4);

  float* outAx = (float*)d_out;
  float* outAn = outAx + NN * 3;
  float* outBx = outAn + (size_t)NN * 128;
  float* outBn = outBx + NN * 3;

  auto convw = [&](const float* s, _Float16* d, int K, int N, int Kpad) {
    int total = N * Kpad;
    convw_kernel<<<(total + 255) / 256, 256, 0, stream>>>(s, d, K, N, Kpad);
  };
  convw(eA_W1, W1A_T, 335, 128, 352);
  convw(eB_W1, W1B_T, 335, 128, 352);
  convw(eA_W2, W2A_T, 128, 128, 128);
  convw(eB_W2, W2B_T, 128, 128, 128);
  convw(cA_W1, CW1A_T, 128, 128, 128);
  convw(cB_W1, CW1B_T, 128, 128, 128);
  convw(WQ, WQ_T, 128, 128, 128);
  convw(WK, WK_T, 128, 128, 128);
  convw(WV, WV_T, 128, 128, 128);
  convw(nA_W1, NW1A_T, 512, 128, 512);
  convw(nB_W1, NW1B_T, 512, 128, 512);
  convw(nA_W2, NW2A_T, 128, 128, 128);
  convw(nB_W2, NW2B_T, 128, 128, 128);

  {
    int zn = 2 * NN * 128 + 2 * NN * 3 + 2 * NN;
    zero_kernel<<<(zn + 255) / 256, 256, 0, stream>>>(aggrA, zn);
  }

  lin_kernel<<<NN / 64, 256, 0, stream>>>(h_A, WQ_T, Qh, 1, 0);
  lin_kernel<<<NN / 64, 256, 0, stream>>>(h_B, WK_T, Kh, 1, 0);
  lin_kernel<<<NN / 64, 256, 0, stream>>>(h_B, WV_T, VT, 0, 1);

  edge_kernel<<<EE / 64, 256, 0, stream>>>(coords_A, h_A, efeat_A, srcA, dstA,
      W1A_T, eA_b1, W2A_T, eA_b2, CW1A_T, cA_b1, cA_W2, cA_b2, aggrA, xupA, cntA);
  edge_kernel<<<EE / 64, 256, 0, stream>>>(coords_B, h_B, efeat_B, srcB, dstB,
      W1B_T, eB_b1, W2B_T, eB_b2, CW1B_T, cB_b1, cB_W2, cB_b2, aggrB, xupB, cntB);

  attn_kernel<<<NN / 128, 256, 0, stream>>>(Qh, Kh, VT, mask, crossA);

  nodemlp_kernel<<<NN / 32, 256, 0, stream>>>(h_A, aggrA, cntA, crossA, 1, ofeat_A,
      NW1A_T, nA_b1, NW2A_T, nA_b2, outAn);
  nodemlp_kernel<<<NN / 32, 256, 0, stream>>>(h_B, aggrB, cntB, crossA, 0, ofeat_B,
      NW1B_T, nB_b1, NW2B_T, nB_b2, outBn);

  coords_kernel<<<(NN * 3 + 255) / 256, 256, 0, stream>>>(ocrd_A, coords_A, xupA, cntA, outAx);
  coords_kernel<<<(NN * 3 + 255) / 256, 256, 0, stream>>>(ocrd_B, coords_B, xupB, cntB, outBx);
}
